// GrAttCPI_68779606278686
// MI455X (gfx1250) — matrix operand order repaired and checked by reading, not compiled, not run on hardware
//
#include <hip/hip_runtime.h>
#include <hip/hip_bf16.h>
#include <math.h>

// ---------------------------------------------------------------------------
// GrAttCPI forward for MI455X (gfx1250, wave32, WMMA).
// GAT projections run as bf16 WMMA GEMMs (v_wmma_f32_16x16x32_bf16), one
// 16x64 strip per wave (A-fragment reused across 4 WMMAs);
// edge softmax/aggregation as wave-per-edge kernels (L2-resident gathers);
// the 734MB fc1_t matvec as an HBM-bound K-chunked f32 kernel w/ prefetch.
// ---------------------------------------------------------------------------

typedef __attribute__((ext_vector_type(16))) __bf16 v16bf;
typedef __attribute__((ext_vector_type(8)))  __bf16 v8bf;
typedef __attribute__((ext_vector_type(8)))  float  v8f;

static constexpr int BATCH = 16;
static constexpr int NTPER = 1400;
static constexpr int NCPER = 148;
static constexpr int NTN   = BATCH * NTPER;   // 22400
static constexpr int NCN   = BATCH * NCPER;   // 2368
static constexpr int ETE   = 268800;
static constexpr int ECE   = 9472;

#define CDIV(a, b) (((a) + (b) - 1) / (b))

// ----------------------------- helpers -------------------------------------

__device__ __forceinline__ void atomicMaxFloat(float* addr, float val) {
  if (val >= 0.0f) atomicMax((int*)addr, __float_as_int(val));
  else             atomicMin((unsigned int*)addr, __float_as_uint(val));
}

// ------------------------- conversion kernels -------------------------------

// dst[r][c] (r<rows, c<dcols) = c<cols ? bf16(src[r][c]) : 0
__global__ void k_cvt_bf16_pad(const float* __restrict__ src, __bf16* __restrict__ dst,
                               int rows, int cols, int dcols) {
  long i = (long)blockIdx.x * blockDim.x + threadIdx.x;
  long total = (long)rows * dcols;
  if (i >= total) return;
  int r = (int)(i / dcols), c = (int)(i % dcols);
  dst[i] = (c < cols) ? (__bf16)src[(size_t)r * cols + c] : (__bf16)0.0f;
}

// src: K x N (row major).  dst: N x Kp, dst[n][k] = k<K ? src[k][n] : 0
__global__ void k_transpose_bf16_pad(const float* __restrict__ src, __bf16* __restrict__ dst,
                                     int K, int N, int Kp) {
  long i = (long)blockIdx.x * blockDim.x + threadIdx.x;
  long total = (long)N * Kp;
  if (i >= total) return;
  int n = (int)(i / Kp), k = (int)(i % Kp);
  dst[i] = (k < K) ? (__bf16)src[(size_t)k * N + n] : (__bf16)0.0f;
}

__global__ void k_fill_f32(float* __restrict__ p, float v, long n) {
  long i = (long)blockIdx.x * blockDim.x + threadIdx.x;
  if (i < n) p[i] = v;
}

// out[r][c] = bias[c]
__global__ void k_rows_bias(float* __restrict__ out, const float* __restrict__ bias,
                            int rows, int cols) {
  long i = (long)blockIdx.x * blockDim.x + threadIdx.x;
  long total = (long)rows * cols;
  if (i >= total) return;
  out[i] = bias[(int)(i % cols)];
}

__global__ void k_elu_cvt(const float* __restrict__ x, __bf16* __restrict__ y, long n) {
  long i = (long)blockIdx.x * blockDim.x + threadIdx.x;
  if (i >= n) return;
  float v = x[i];
  y[i] = (__bf16)((v > 0.0f) ? v : (__expf(v) - 1.0f));
}

// x laid out [BATCH*nper, d]; zero rows >= cnt[b] within each batch chunk
__global__ void k_mask_flatten(const float* __restrict__ x, const int* __restrict__ cnt,
                               float* __restrict__ out, int nper, int d) {
  long i = (long)blockIdx.x * blockDim.x + threadIdx.x;
  long total = (long)BATCH * nper * d;
  if (i >= total) return;
  int b = (int)(i / ((long)nper * d));
  int r = (int)((i / d) % nper);
  out[i] = (r < cnt[b]) ? x[i] : 0.0f;
}

// bn in place: x = g*(x-m)*rsqrt(v+1e-5) + b
__global__ void k_bn(float* __restrict__ x, const float* __restrict__ g,
                     const float* __restrict__ b, const float* __restrict__ m,
                     const float* __restrict__ v, int rows, int cols) {
  long i = (long)blockIdx.x * blockDim.x + threadIdx.x;
  long total = (long)rows * cols;
  if (i >= total) return;
  int c = (int)(i % cols);
  x[i] = g[c] * (x[i] - m[c]) * rsqrtf(v[c] + 1e-5f) + b[c];
}

// ------------------------------ WMMA GEMM -----------------------------------
// C[M,N] = A[M,K](bf16) x Bt[N,K](bf16, pre-transposed weight) + bias[N]
// One wave computes one 16x64 strip (4 x 16x16 tiles, one shared A fragment).
// N must be a multiple of 64 (512 and 64 here), K a multiple of 32.
// A lane layout (matches B):  m = lane%16,
//   contiguous [koff, koff+16), koff = (lane<16 ? 0 : 16).
// B lane layout (same contiguous k order as A, not AMD's table order): n = lane%16,
//   contiguous [boff, boff+16), boff = (lane<16 ? 0 : 16).
__global__ __launch_bounds__(256) void k_wmma_gemm_bf16(
    const __bf16* __restrict__ A, const __bf16* __restrict__ Bt,
    const float* __restrict__ bias, float* __restrict__ C,
    int M, int N, int K) {
  int lane = threadIdx.x & 31;
  int wave = threadIdx.x >> 5;
  int strips_n = N >> 6;
  int total = (M >> 4) * strips_n;
  int strip = blockIdx.x * 8 + wave;
  if (strip >= total) return;                // wave-uniform: EXEC stays all-1s
  int tm = strip / strips_n, tn = strip % strips_n;
  int lrow = lane & 15;
  int row  = (tm << 4) + lrow;
  int col0 = (tn << 6) + lrow;               // lane's column in sub-tile 0
  int koff = (lane < 16) ? 0 : 16;
  int boff = (lane < 16) ? 0 : 16;
  const __bf16* arow = A  + (size_t)row  * K + koff;
  const __bf16* brow = Bt + (size_t)col0 * K + boff;
  size_t bstep = (size_t)16 * K;             // 16 columns of Bt
  v8f acc[4] = {};
  for (int k = 0; k < K; k += 32) {
    v8bf alo = *(const v8bf*)(arow + k);
    v8bf ahi = *(const v8bf*)(arow + k + 8);
    v16bf a;
#pragma unroll
    for (int i = 0; i < 8; ++i) { a[i] = alo[i]; a[i + 8] = ahi[i]; }
    v16bf b0 = *(const v16bf*)(brow + k);
    v16bf b1 = *(const v16bf*)(brow + bstep + k);
    v16bf b2 = *(const v16bf*)(brow + 2 * bstep + k);
    v16bf b3 = *(const v16bf*)(brow + 3 * bstep + k);
    acc[0] = __builtin_amdgcn_wmma_f32_16x16x32_bf16(false, a, false, b0, (short)0, acc[0], false, false);
    acc[1] = __builtin_amdgcn_wmma_f32_16x16x32_bf16(false, a, false, b1, (short)0, acc[1], false, false);
    acc[2] = __builtin_amdgcn_wmma_f32_16x16x32_bf16(false, a, false, b2, (short)0, acc[2], false, false);
    acc[3] = __builtin_amdgcn_wmma_f32_16x16x32_bf16(false, a, false, b3, (short)0, acc[3], false, false);
  }
  int r0 = (tm << 4) + ((lane < 16) ? 0 : 8);
#pragma unroll
  for (int j = 0; j < 4; ++j) {
    int col = col0 + 16 * j;
    float bv = bias ? bias[col] : 0.0f;
#pragma unroll
    for (int i = 0; i < 8; ++i)
      C[(size_t)(r0 + i) * N + col] = acc[j][i] + bv;
  }
}

// --------------------------- GATv2 edge kernels -----------------------------
// H heads of C channels; HC = H*C in {512, 64}. One wave per edge.

template <int H, int C>
__global__ __launch_bounds__(256) void k_edge_logits_max(
    const float* __restrict__ xl, const float* __restrict__ xr,
    const float* __restrict__ We, const float* __restrict__ attr,
    const float* __restrict__ att, const int* __restrict__ src,
    const int* __restrict__ dst, float* __restrict__ logits,
    float* __restrict__ lmax, int E) {
  constexpr int HC  = H * C;
  constexpr int EPL = HC / 32;   // elements per lane
  constexpr int R   = 32 / H;    // lanes per head
  int lane = threadIdx.x & 31;
  int e = blockIdx.x * (blockDim.x >> 5) + (threadIdx.x >> 5);
  if (e >= E) return;
  int s = src[e], d = dst[e];
  int h = lane / R;
  int base = h * C + (lane % R) * EPL;
  float av = attr ? attr[e] : 0.0f;
  float part = 0.0f;
#pragma unroll
  for (int i = 0; i < EPL; ++i) {
    int j = base + i;
    float x = xl[(size_t)s * HC + j] + xr[(size_t)d * HC + j];
    if (We) x += av * We[j];
    x = (x > 0.0f) ? x : 0.2f * x;           // leaky_relu(0.2)
    part += x * att[j];
  }
#pragma unroll
  for (int off = R >> 1; off > 0; off >>= 1)
    part += __shfl_xor(part, off, 32);
  if ((lane % R) == 0) {
    logits[(size_t)e * H + h] = part;
    atomicMaxFloat(&lmax[(size_t)d * H + h], part);
  }
}

template <int H>
__global__ void k_edge_exp_sum(const int* __restrict__ dst, float* __restrict__ logits,
                               const float* __restrict__ lmax, float* __restrict__ denom,
                               int E) {
  long i = (long)blockIdx.x * blockDim.x + threadIdx.x;
  if (i >= (long)E * H) return;
  int e = (int)(i / H), h = (int)(i % H);
  int d = dst[e];
  float a = __expf(logits[i] - lmax[(size_t)d * H + h]);
  logits[i] = a;                              // overwrite with exp value
  atomicAdd(&denom[(size_t)d * H + h], a);
}

template <int H, int C>
__global__ __launch_bounds__(256) void k_edge_scatter(
    const float* __restrict__ xl, const int* __restrict__ src,
    const int* __restrict__ dst, const float* __restrict__ aexp,
    const float* __restrict__ denom, float* __restrict__ out, int E) {
  constexpr int HC  = H * C;
  constexpr int EPL = HC / 32;
  constexpr int R   = 32 / H;
  int lane = threadIdx.x & 31;
  int e = blockIdx.x * (blockDim.x >> 5) + (threadIdx.x >> 5);
  if (e >= E) return;
  int s = src[e], d = dst[e];
  int h = lane / R;
  int base = h * C + (lane % R) * EPL;
  float alpha = aexp[(size_t)e * H + h] / (denom[(size_t)d * H + h] + 1e-16f);
#pragma unroll
  for (int i = 0; i < EPL; ++i) {
    int j = base + i;
    atomicAdd(&out[(size_t)d * HC + j], alpha * xl[(size_t)s * HC + j]);
  }
}

// -------------------------- 16-row FC (matvec x16) --------------------------
// C[16,N] += A[16,K] x W[K,N]; K split over blockIdx.y chunks of 896 (LDS staged).
static constexpr int KCH = 896;

__global__ __launch_bounds__(128) void k_fc16(
    const float* __restrict__ A, const float* __restrict__ W,
    float* __restrict__ C, int K, int N) {
  __shared__ float sA[16 * KCH];
  int k0 = blockIdx.y * KCH;
  int kn = K - k0; if (kn > KCH) kn = KCH;
  for (int i = threadIdx.x; i < 16 * KCH; i += 128) {
    int bb = i / KCH, kk = i % KCH;
    sA[i] = (kk < kn) ? A[(size_t)bb * K + k0 + kk] : 0.0f;
  }
  __syncthreads();
  int n = blockIdx.x * 128 + threadIdx.x;
  if (n >= N) return;
  float acc[16];
#pragma unroll
  for (int bb = 0; bb < 16; ++bb) acc[bb] = 0.0f;
  for (int kk = 0; kk < kn; ++kk) {
    // stay ahead of the 23.3 TB/s weight stream (global_prefetch_b8)
    if (kk + 8 < kn) __builtin_prefetch(&W[(size_t)(k0 + kk + 8) * N + n], 0, 1);
    float w = W[(size_t)(k0 + kk) * N + n];
#pragma unroll
    for (int bb = 0; bb < 16; ++bb) acc[bb] += sA[bb * KCH + kk] * w;
  }
#pragma unroll
  for (int bb = 0; bb < 16; ++bb) atomicAdd(&C[(size_t)bb * N + n], acc[bb]);
}

// ------------------------------- tail MLP -----------------------------------
// concat([c,t]) [16,256] -> fc1/bn1 -> fc2/bn2 -> fc3/bn3 -> fc4 -> out[16,2]
__global__ __launch_bounds__(256) void k_tail(
    const float* __restrict__ c128, const float* __restrict__ t128,
    const float* __restrict__ W1, const float* __restrict__ b1,
    const float* __restrict__ g1, const float* __restrict__ be1,
    const float* __restrict__ m1, const float* __restrict__ v1,
    const float* __restrict__ W2, const float* __restrict__ b2,
    const float* __restrict__ g2, const float* __restrict__ be2,
    const float* __restrict__ m2, const float* __restrict__ v2,
    const float* __restrict__ W3, const float* __restrict__ b3,
    const float* __restrict__ g3, const float* __restrict__ be3,
    const float* __restrict__ m3, const float* __restrict__ v3,
    const float* __restrict__ W4, const float* __restrict__ b4,
    float* __restrict__ out) {
  __shared__ float r0[16 * 256];
  __shared__ float r1[16 * 64];
  __shared__ float r2[16 * 16];
  __shared__ float r3[16 * 4];
  int tid = threadIdx.x;
  for (int i = tid; i < 16 * 256; i += 256) {
    int b = i >> 8, j = i & 255;
    r0[i] = (j < 128) ? c128[b * 128 + j] : t128[b * 128 + j - 128];
  }
  __syncthreads();
  for (int i = tid; i < 16 * 64; i += 256) {
    int b = i >> 6, j = i & 63;
    float acc = b1[j];
    for (int k = 0; k < 256; ++k) acc += r0[b * 256 + k] * W1[k * 64 + j];
    r1[i] = g1[j] * (acc - m1[j]) * rsqrtf(v1[j] + 1e-5f) + be1[j];
  }
  __syncthreads();
  for (int i = tid; i < 16 * 16; i += 256) {
    int b = i >> 4, j = i & 15;
    float acc = b2[j];
    for (int k = 0; k < 64; ++k) acc += r1[b * 64 + k] * W2[k * 16 + j];
    r2[i] = g2[j] * (acc - m2[j]) * rsqrtf(v2[j] + 1e-5f) + be2[j];
  }
  __syncthreads();
  for (int i = tid; i < 16 * 4; i += 256) {
    int b = i >> 2, j = i & 3;
    float acc = b3[j];
    for (int k = 0; k < 16; ++k) acc += r2[b * 16 + k] * W3[k * 4 + j];
    r3[i] = g3[j] * (acc - m3[j]) * rsqrtf(v3[j] + 1e-5f) + be3[j];
  }
  __syncthreads();
  for (int i = tid; i < 32; i += 256) {
    int b = i >> 1, j = i & 1;
    float acc = b4[j];
    for (int k = 0; k < 4; ++k) acc += r3[b * 4 + k] * W4[k * 2 + j];
    out[i] = acc;
  }
}

// ------------------------------- host side ----------------------------------

extern "C" void kernel_launch(void* const* d_in, const int* in_sizes, int n_in,
                              void* d_out, int out_size, void* d_ws, size_t ws_size,
                              hipStream_t stream) {
  (void)in_sizes; (void)n_in; (void)out_size; (void)ws_size;

  const float* tx     = (const float*)d_in[0];
  const int*   t_ei   = (const int*)d_in[1];
  const float* t_attr = (const float*)d_in[2];
  const int*   naas   = (const int*)d_in[3];
  const float* dx     = (const float*)d_in[4];
  const int*   c_ei   = (const int*)d_in[5];
  const int*   natoms = (const int*)d_in[6];
  const float* const* pp = (const float* const*)(d_in + 7);
  // params flattened in sorted-pytree order (jax dict => sorted keys):
  const float *bn1_b = pp[0],  *bn1_g = pp[1],  *bn1_m = pp[2],  *bn1_v = pp[3];
  const float *bn1t_b = pp[4], *bn1t_g = pp[5], *bn1t_m = pp[6], *bn1t_v = pp[7];
  const float *bn2_b = pp[8],  *bn2_g = pp[9],  *bn2_m = pp[10], *bn2_v = pp[11];
  const float *bn2t_b = pp[12],*bn2t_g = pp[13],*bn2t_m = pp[14],*bn2t_v = pp[15];
  const float *bn3_b = pp[16], *bn3_g = pp[17], *bn3_m = pp[18], *bn3_v = pp[19];
  const float *bn3t_b = pp[20],*bn3t_g = pp[21],*bn3t_m = pp[22],*bn3t_v = pp[23];
  const float *bnc_b = pp[24], *bnc_g = pp[25], *bnc_m = pp[26], *bnc_v = pp[27];
  const float *fc1_W = pp[28], *fc1_b = pp[29];
  const float *fc1t_W = pp[30],*fc1t_b = pp[31];
  const float *fc2_W = pp[32], *fc2_b = pp[33];
  const float *fc2t_W = pp[34],*fc2t_b = pp[35];
  const float *fc3_W = pp[36], *fc3_b = pp[37];
  const float *fc3t_W = pp[38],*fc3t_b = pp[39];
  const float *fc4_W = pp[40], *fc4_b = pp[41];
  const float *fcc_W = pp[42], *fcc_b = pp[43];
  const float *g1c_Wl = pp[44],*g1c_Wr = pp[45],*g1c_att = pp[46],*g1c_bias = pp[47],*g1c_bl = pp[48],*g1c_br = pp[49];
  const float *g1t_We = pp[50],*g1t_Wl = pp[51],*g1t_Wr = pp[52],*g1t_att = pp[53],*g1t_bias = pp[54],*g1t_bl = pp[55],*g1t_br = pp[56];
  const float *g2c_Wl = pp[57],*g2c_Wr = pp[58],*g2c_att = pp[59],*g2c_bias = pp[60],*g2c_bl = pp[61],*g2c_br = pp[62];
  const float *g2t_We = pp[63],*g2t_Wl = pp[64],*g2t_Wr = pp[65],*g2t_att = pp[66],*g2t_bias = pp[67],*g2t_bl = pp[68],*g2t_br = pp[69];

  const int* t_src = t_ei;            const int* t_dst = t_ei + ETE;
  const int* c_src = c_ei;            const int* c_dst = c_ei + ECE;

  // ----- workspace carve-out (256B aligned) -----
  size_t off = 0;
  auto wsa = [&](size_t bytes) -> void* {
    void* p = (char*)d_ws + off;
    off += (bytes + 255) & ~(size_t)255;
    return p;
  };
  __bf16* tx_bf  = (__bf16*)wsa((size_t)NTN * 1280 * 2);
  __bf16* WlT1t  = (__bf16*)wsa((size_t)512 * 1280 * 2);
  __bf16* WrT1t  = (__bf16*)wsa((size_t)512 * 1280 * 2);
  float*  xl1t   = (float*)wsa((size_t)NTN * 512 * 4);
  float*  xr1t   = (float*)wsa((size_t)NTN * 512 * 4);
  float*  out1t  = (float*)wsa((size_t)NTN * 512 * 4);
  float*  lg1t   = (float*)wsa((size_t)ETE * 8 * 4);
  float*  mx1t   = (float*)wsa((size_t)NTN * 8 * 4);
  float*  dn1t   = (float*)wsa((size_t)NTN * 8 * 4);
  __bf16* t1_bf  = (__bf16*)wsa((size_t)NTN * 512 * 2);
  __bf16* WlT2t  = (__bf16*)wsa((size_t)64 * 512 * 2);
  __bf16* WrT2t  = (__bf16*)wsa((size_t)64 * 512 * 2);
  float*  xl2t   = (float*)wsa((size_t)NTN * 64 * 4);
  float*  xr2t   = (float*)wsa((size_t)NTN * 64 * 4);
  float*  out2t  = (float*)wsa((size_t)NTN * 64 * 4);
  float*  lg2t   = (float*)wsa((size_t)ETE * 4);
  float*  mx2t   = (float*)wsa((size_t)NTN * 4);
  float*  dn2t   = (float*)wsa((size_t)NTN * 4);
  float*  mt     = (float*)wsa((size_t)16 * 89600 * 4);
  float*  f1t    = (float*)wsa((size_t)16 * 2048 * 4);
  float*  f2t    = (float*)wsa((size_t)16 * 512 * 4);
  float*  f3t    = (float*)wsa((size_t)16 * 128 * 4);
  __bf16* dx_bf  = (__bf16*)wsa((size_t)NCN * 96 * 2);
  __bf16* WlT1c  = (__bf16*)wsa((size_t)512 * 96 * 2);
  __bf16* WrT1c  = (__bf16*)wsa((size_t)512 * 96 * 2);
  float*  xl1c   = (float*)wsa((size_t)NCN * 512 * 4);
  float*  xr1c   = (float*)wsa((size_t)NCN * 512 * 4);
  float*  out1c  = (float*)wsa((size_t)NCN * 512 * 4);
  float*  lg1c   = (float*)wsa((size_t)ECE * 8 * 4);
  float*  mx1c   = (float*)wsa((size_t)NCN * 8 * 4);
  float*  dn1c   = (float*)wsa((size_t)NCN * 8 * 4);
  __bf16* c1_bf  = (__bf16*)wsa((size_t)NCN * 512 * 2);
  __bf16* WlT2c  = (__bf16*)wsa((size_t)64 * 512 * 2);
  __bf16* WrT2c  = (__bf16*)wsa((size_t)64 * 512 * 2);
  float*  xl2c   = (float*)wsa((size_t)NCN * 64 * 4);
  float*  xr2c   = (float*)wsa((size_t)NCN * 64 * 4);
  float*  out2c  = (float*)wsa((size_t)NCN * 64 * 4);
  float*  lg2c   = (float*)wsa((size_t)ECE * 4);
  float*  mx2c   = (float*)wsa((size_t)NCN * 4);
  float*  dn2c   = (float*)wsa((size_t)NCN * 4);
  float*  mc     = (float*)wsa((size_t)16 * 9472 * 4);
  float*  fcp    = (float*)wsa((size_t)16 * 128 * 4);

  const float NEG_INF = -HUGE_VALF;

  auto gemm = [&](const __bf16* A, const __bf16* Bt, const float* bias, float* C,
                  int M, int N, int K) {
    int strips = (M / 16) * (N / 64);
    k_wmma_gemm_bf16<<<CDIV(strips, 8), 256, 0, stream>>>(A, Bt, bias, C, M, N, K);
  };

  // ----- conversions -----
  k_cvt_bf16_pad<<<CDIV((long)NTN * 1280, 256), 256, 0, stream>>>(tx, tx_bf, NTN, 1280, 1280);
  k_transpose_bf16_pad<<<CDIV(512 * 1280, 256), 256, 0, stream>>>(g1t_Wl, WlT1t, 1280, 512, 1280);
  k_transpose_bf16_pad<<<CDIV(512 * 1280, 256), 256, 0, stream>>>(g1t_Wr, WrT1t, 1280, 512, 1280);
  k_transpose_bf16_pad<<<CDIV(64 * 512, 256), 256, 0, stream>>>(g2t_Wl, WlT2t, 512, 64, 512);
  k_transpose_bf16_pad<<<CDIV(64 * 512, 256), 256, 0, stream>>>(g2t_Wr, WrT2t, 512, 64, 512);
  k_cvt_bf16_pad<<<CDIV((long)NCN * 96, 256), 256, 0, stream>>>(dx, dx_bf, NCN, 78, 96);
  k_transpose_bf16_pad<<<CDIV(512 * 96, 256), 256, 0, stream>>>(g1c_Wl, WlT1c, 78, 512, 96);
  k_transpose_bf16_pad<<<CDIV(512 * 96, 256), 256, 0, stream>>>(g1c_Wr, WrT1c, 78, 512, 96);
  k_transpose_bf16_pad<<<CDIV(64 * 512, 256), 256, 0, stream>>>(g2c_Wl, WlT2c, 512, 64, 512);
  k_transpose_bf16_pad<<<CDIV(64 * 512, 256), 256, 0, stream>>>(g2c_Wr, WrT2c, 512, 64, 512);

  // ===== protein branch =====
  gemm(tx_bf, WlT1t, g1t_bl, xl1t, NTN, 512, 1280);
  gemm(tx_bf, WrT1t, g1t_br, xr1t, NTN, 512, 1280);

  k_fill_f32<<<CDIV((long)NTN * 8, 256), 256, 0, stream>>>(mx1t, NEG_INF, (long)NTN * 8);
  k_fill_f32<<<CDIV((long)NTN * 8, 256), 256, 0, stream>>>(dn1t, 0.0f, (long)NTN * 8);
  k_rows_bias<<<CDIV((long)NTN * 512, 256), 256, 0, stream>>>(out1t, g1t_bias, NTN, 512);
  k_edge_logits_max<8, 64><<<CDIV(ETE, 8), 256, 0, stream>>>(
      xl1t, xr1t, g1t_We, t_attr, g1t_att, t_src, t_dst, lg1t, mx1t, ETE);
  k_edge_exp_sum<8><<<CDIV((long)ETE * 8, 256), 256, 0, stream>>>(t_dst, lg1t, mx1t, dn1t, ETE);
  k_edge_scatter<8, 64><<<CDIV(ETE, 8), 256, 0, stream>>>(
      xl1t, t_src, t_dst, lg1t, dn1t, out1t, ETE);

  k_elu_cvt<<<CDIV((long)NTN * 512, 256), 256, 0, stream>>>(out1t, t1_bf, (long)NTN * 512);
  gemm(t1_bf, WlT2t, g2t_bl, xl2t, NTN, 64, 512);
  gemm(t1_bf, WrT2t, g2t_br, xr2t, NTN, 64, 512);

  k_fill_f32<<<CDIV(NTN, 256), 256, 0, stream>>>(mx2t, NEG_INF, NTN);
  k_fill_f32<<<CDIV(NTN, 256), 256, 0, stream>>>(dn2t, 0.0f, NTN);
  k_rows_bias<<<CDIV((long)NTN * 64, 256), 256, 0, stream>>>(out2t, g2t_bias, NTN, 64);
  k_edge_logits_max<1, 64><<<CDIV(ETE, 8), 256, 0, stream>>>(
      xl2t, xr2t, g2t_We, t_attr, g2t_att, t_src, t_dst, lg2t, mx2t, ETE);
  k_edge_exp_sum<1><<<CDIV(ETE, 256), 256, 0, stream>>>(t_dst, lg2t, mx2t, dn2t, ETE);
  k_edge_scatter<1, 64><<<CDIV(ETE, 8), 256, 0, stream>>>(
      xl2t, t_src, t_dst, lg2t, dn2t, out2t, ETE);

  k_mask_flatten<<<CDIV((long)16 * 89600, 256), 256, 0, stream>>>(out2t, naas, mt, NTPER, 64);

  k_rows_bias<<<CDIV(16 * 2048, 256), 256, 0, stream>>>(f1t, fc1t_b, 16, 2048);
  { dim3 g(CDIV(2048, 128), CDIV(89600, KCH));
    k_fc16<<<g, 128, 0, stream>>>(mt, fc1t_W, f1t, 89600, 2048); }
  k_bn<<<CDIV(16 * 2048, 256), 256, 0, stream>>>(f1t, bn1t_g, bn1t_b, bn1t_m, bn1t_v, 16, 2048);

  k_rows_bias<<<CDIV(16 * 512, 256), 256, 0, stream>>>(f2t, fc2t_b, 16, 512);
  { dim3 g(CDIV(512, 128), CDIV(2048, KCH));
    k_fc16<<<g, 128, 0, stream>>>(f1t, fc2t_W, f2t, 2048, 512); }
  k_bn<<<CDIV(16 * 512, 256), 256, 0, stream>>>(f2t, bn2t_g, bn2t_b, bn2t_m, bn2t_v, 16, 512);

  k_rows_bias<<<CDIV(16 * 128, 256), 256, 0, stream>>>(f3t, fc3t_b, 16, 128);
  { dim3 g(CDIV(128, 128), CDIV(512, KCH));
    k_fc16<<<g, 128, 0, stream>>>(f2t, fc3t_W, f3t, 512, 128); }
  k_bn<<<CDIV(16 * 128, 256), 256, 0, stream>>>(f3t, bn3t_g, bn3t_b, bn3t_m, bn3t_v, 16, 128);

  // ===== compound branch =====
  gemm(dx_bf, WlT1c, g1c_bl, xl1c, NCN, 512, 96);
  gemm(dx_bf, WrT1c, g1c_br, xr1c, NCN, 512, 96);

  k_fill_f32<<<CDIV((long)NCN * 8, 256), 256, 0, stream>>>(mx1c, NEG_INF, (long)NCN * 8);
  k_fill_f32<<<CDIV((long)NCN * 8, 256), 256, 0, stream>>>(dn1c, 0.0f, (long)NCN * 8);
  k_rows_bias<<<CDIV((long)NCN * 512, 256), 256, 0, stream>>>(out1c, g1c_bias, NCN, 512);
  k_edge_logits_max<8, 64><<<CDIV(ECE, 8), 256, 0, stream>>>(
      xl1c, xr1c, nullptr, nullptr, g1c_att, c_src, c_dst, lg1c, mx1c, ECE);
  k_edge_exp_sum<8><<<CDIV((long)ECE * 8, 256), 256, 0, stream>>>(c_dst, lg1c, mx1c, dn1c, ECE);
  k_edge_scatter<8, 64><<<CDIV(ECE, 8), 256, 0, stream>>>(
      xl1c, c_src, c_dst, lg1c, dn1c, out1c, ECE);

  k_elu_cvt<<<CDIV((long)NCN * 512, 256), 256, 0, stream>>>(out1c, c1_bf, (long)NCN * 512);
  gemm(c1_bf, WlT2c, g2c_bl, xl2c, NCN, 64, 512);
  gemm(c1_bf, WrT2c, g2c_br, xr2c, NCN, 64, 512);

  k_fill_f32<<<CDIV(NCN, 256), 256, 0, stream>>>(mx2c, NEG_INF, NCN);
  k_fill_f32<<<CDIV(NCN, 256), 256, 0, stream>>>(dn2c, 0.0f, NCN);
  k_rows_bias<<<CDIV((long)NCN * 64, 256), 256, 0, stream>>>(out2c, g2c_bias, NCN, 64);
  k_edge_logits_max<1, 64><<<CDIV(ECE, 8), 256, 0, stream>>>(
      xl2c, xr2c, nullptr, nullptr, g2c_att, c_src, c_dst, lg2c, mx2c, ECE);
  k_edge_exp_sum<1><<<CDIV(ECE, 256), 256, 0, stream>>>(c_dst, lg2c, mx2c, dn2c, ECE);
  k_edge_scatter<1, 64><<<CDIV(ECE, 8), 256, 0, stream>>>(
      xl2c, c_src, c_dst, lg2c, dn2c, out2c, ECE);

  k_mask_flatten<<<CDIV((long)16 * 9472, 256), 256, 0, stream>>>(out2c, natoms, mc, NCPER, 64);

  k_rows_bias<<<CDIV(16 * 128, 256), 256, 0, stream>>>(fcp, fcc_b, 16, 128);
  { dim3 g(CDIV(128, 128), CDIV(9472, KCH));
    k_fc16<<<g, 128, 0, stream>>>(mc, fcc_W, fcp, 9472, 128); }
  k_bn<<<CDIV(16 * 128, 256), 256, 0, stream>>>(fcp, bnc_g, bnc_b, bnc_m, bnc_v, 16, 128);

  // ===== fused tail =====
  k_tail<<<1, 256, 0, stream>>>(fcp, f3t,
      fc1_W, fc1_b, bn1_g, bn1_b, bn1_m, bn1_v,
      fc2_W, fc2_b, bn2_g, bn2_b, bn2_m, bn2_v,
      fc3_W, fc3_b, bn3_g, bn3_b, bn3_m, bn3_v,
      fc4_W, fc4_b, (float*)d_out);
}
